// GraphSAGE_80573586473556
// MI455X (gfx1250) — compile-verified, matrix operand order checked by reading, not run on hardware
//
#include <hip/hip_runtime.h>
#include <hip/hip_bf16.h>

// ---------------- types ----------------
typedef __attribute__((ext_vector_type(16))) __bf16 v16bf;
typedef __attribute__((ext_vector_type(8)))  __bf16 v8bf;
typedef __attribute__((ext_vector_type(8)))  float  v8f;

#define N_NODES 100000
#define N_EDGES 1600000
#define IN_F    128
#define HID_F   256
#define OUT_F   128

// ---------------- bf16 helpers (round-to-nearest-even) ----------------
__device__ __forceinline__ unsigned short f2bf_u(float f) {
  unsigned u = __builtin_bit_cast(unsigned, f);
  u += 0x7FFFu + ((u >> 16) & 1u);
  return (unsigned short)(u >> 16);
}
__device__ __forceinline__ __bf16 f2bf(float f) {
  return __builtin_bit_cast(__bf16, f2bf_u(f));
}
__device__ __forceinline__ float bf2f(__bf16 h) {
  unsigned u = ((unsigned)__builtin_bit_cast(unsigned short, h)) << 16;
  return __builtin_bit_cast(float, u);
}

// ---------------- degree: deg[dst] += 1 ----------------
__global__ void k_degree(const int* __restrict__ dst, float* __restrict__ deg) {
  int e = blockIdx.x * blockDim.x + threadIdx.x;
  if (e < N_EDGES) unsafeAtomicAdd(&deg[dst[e]], 1.0f);
}

// ---------------- f32 -> bf16 convert ----------------
__global__ void k_f32_to_bf16(const float* __restrict__ in, __bf16* __restrict__ out, int n) {
  int i = blockIdx.x * blockDim.x + threadIdx.x;
  if (i < n) out[i] = f2bf(in[i]);
}

// ---------------- scatter-add of f32 features: agg[dst] += x[src] ----------------
// one thread per (edge, 4-float chunk); lgc = log2(feat/4)
__global__ void k_scatter_f32(const float* __restrict__ x, const int* __restrict__ src,
                              const int* __restrict__ dst, float* __restrict__ agg,
                              int lgc, int feat) {
  long long idx = (long long)blockIdx.x * blockDim.x + threadIdx.x;
  int e = (int)(idx >> lgc);
  if (e >= N_EDGES) return;
  int c = (((int)idx) & ((1 << lgc) - 1)) << 2;
  int s = src[e], d = dst[e];
  const float4 v = *(const float4*)(x + (size_t)s * feat + c);
  float* o = agg + (size_t)d * feat + c;
  unsafeAtomicAdd(o + 0, v.x);
  unsafeAtomicAdd(o + 1, v.y);
  unsafeAtomicAdd(o + 2, v.z);
  unsafeAtomicAdd(o + 3, v.w);
}

// ---------------- scatter-add of bf16 features (layer-2 gather of h) ----------------
__global__ void k_scatter_bf16(const __bf16* __restrict__ h, const int* __restrict__ src,
                               const int* __restrict__ dst, float* __restrict__ agg,
                               int lgc, int feat) {
  long long idx = (long long)blockIdx.x * blockDim.x + threadIdx.x;
  int e = (int)(idx >> lgc);
  if (e >= N_EDGES) return;
  int c = (((int)idx) & ((1 << lgc) - 1)) << 2;
  int s = src[e], d = dst[e];
  const __bf16* p = h + (size_t)s * feat + c;
  float* o = agg + (size_t)d * feat + c;
  unsafeAtomicAdd(o + 0, bf2f(p[0]));
  unsafeAtomicAdd(o + 1, bf2f(p[1]));
  unsafeAtomicAdd(o + 2, bf2f(p[2]));
  unsafeAtomicAdd(o + 3, bf2f(p[3]));
}

// ---------------- neighbor mean (agg/deg, 0 if deg==0) -> bf16 ----------------
__global__ void k_neigh_mean(const float* __restrict__ agg, const float* __restrict__ deg,
                             __bf16* __restrict__ out, int feat_shift, long long total) {
  long long i = (long long)blockIdx.x * blockDim.x + threadIdx.x;
  if (i >= total) return;
  int row = (int)(i >> feat_shift);
  float d = deg[row];
  float s = agg[i];
  out[i] = f2bf(d > 0.f ? s / d : 0.f);
}

// ---------------- pack K-concatenated weights into WMMA B-matrix lane layout ----------
// B tile (32x16, KxN), bf16: lanes 0-15 -> N=lane,  K in {0..7, 16..23}
//                           lanes 16-31 -> N=lane-16, K in {8..15, 24..31}
// linear t = ((kt*ntiles + ntile)*32 + lane)*16 + j
__global__ void k_pack_w(const float* __restrict__ W0, const float* __restrict__ W1,
                         int K0, int Ktot, int Nw, __bf16* __restrict__ out) {
  int t = blockIdx.x * blockDim.x + threadIdx.x;
  if (t >= Ktot * Nw) return;
  int j    = t & 15;
  int lane = (t >> 4) & 31;
  int tmp  = t >> 9;
  int ntiles = Nw >> 4;
  int ntile = tmp % ntiles;
  int kt    = tmp / ntiles;
  int base  = (j < 8) ? ((lane < 16) ? 0 : 8) : ((lane < 16) ? 16 : 24);
  int k = kt * 32 + base + (j & 7);
  int n = ntile * 16 + (lane & 15);
  float v = (k < K0) ? W0[(size_t)k * Nw + n] : W1[(size_t)(k - K0) * Nw + n];
  out[t] = f2bf(v);
}

// ---------------- fused SAGE GEMM: C = [A0|A1] @ Wpack + bias, optional relu --------
// Each wave computes NT adjacent 16x16 C tiles: A is loaded once per K-step and fed to
// NT back-to-back v_wmma_f32_16x16x32_bf16 ops (register-level A reuse, 4x less A traffic).
template <int NT>
__global__ void k_sage_gemm(const __bf16* __restrict__ A0, const __bf16* __restrict__ A1,
                            int strideA, int K0, int Ktot,
                            const __bf16* __restrict__ Wp, const float* __restrict__ bias,
                            int Nw, float* __restrict__ outF, __bf16* __restrict__ outB,
                            int relu) {
  const int lane  = threadIdx.x & 31;
  const int wave  = threadIdx.x >> 5;
  const int mtile = blockIdx.x;
  const int ntiles = Nw >> 4;
  const int ntile0 = (blockIdx.y * (blockDim.x >> 5) + wave) * NT;
  const int row = mtile * 16 + (lane & 15);     // A-matrix: lanes 0-15 & 16-31 both hold M=0..15
  const int b0  = (lane < 16) ? 0 : 8;          // K sub-block per lane half

  v8f acc[NT];
#pragma unroll
  for (int t = 0; t < NT; ++t) acc[t] = (v8f){};

  const int nkt = Ktot >> 5;
  for (int kt = 0; kt < nkt; ++kt) {
    const int kb = kt << 5;
    const __bf16* Ap = (kb < K0) ? (A0 + (size_t)row * strideA + kb)
                                 : (A1 + (size_t)row * strideA + (kb - K0));
    // per-lane A: K = b0..b0+7 and b0+16..b0+23  (two 16B aligned chunks)
    v8bf alo = *(const v8bf*)(Ap + b0);
    v8bf ahi = *(const v8bf*)(Ap + b0 + 16);
    v16bf a = __builtin_shufflevector(alo, ahi, 0, 1, 2, 3, 4, 5, 6, 7,
                                      8, 9, 10, 11, 12, 13, 14, 15);
    // pre-packed B: contiguous 16 bf16 per lane, adjacent ntiles 512 elems apart
    const __bf16* bp = Wp + (((size_t)kt * ntiles + ntile0) * 32 + lane) * 16;
#pragma unroll
    for (int t = 0; t < NT; ++t) {
      v16bf b = *(const v16bf*)(bp + (size_t)t * (32 * 16));
      acc[t] = __builtin_amdgcn_wmma_f32_16x16x32_bf16(false, a, false, b,
                                                       (short)0, acc[t], false, false);
    }
  }

  // C/D layout: VGPR r -> row (r + b0), col = lane&15
  const int rbase = mtile * 16 + b0;
#pragma unroll
  for (int t = 0; t < NT; ++t) {
    const int col = (ntile0 + t) * 16 + (lane & 15);
    const float bv = bias[col];
#pragma unroll
    for (int r = 0; r < 8; ++r) {
      float v = acc[t][r] + bv;
      if (relu) v = fmaxf(v, 0.f);
      if (outF) outF[(size_t)(rbase + r) * Nw + col] = v;
      else      outB[(size_t)(rbase + r) * Nw + col] = f2bf(v);
    }
  }
}

// ---------------- host orchestration ----------------
extern "C" void kernel_launch(void* const* d_in, const int* in_sizes, int n_in,
                              void* d_out, int out_size, void* d_ws, size_t ws_size,
                              hipStream_t stream) {
  const float* x        = (const float*)d_in[0];
  const int*   src      = (const int*)  d_in[1];
  const int*   dst      = (const int*)  d_in[2];
  const float* W_self1  = (const float*)d_in[3];
  const float* W_neigh1 = (const float*)d_in[4];
  const float* b1       = (const float*)d_in[5];
  const float* W_self2  = (const float*)d_in[6];
  const float* W_neigh2 = (const float*)d_in[7];
  const float* b2       = (const float*)d_in[8];
  float* out = (float*)d_out;

  char* ws = (char*)d_ws;
  size_t off = 0;
  auto alloc = [&](size_t bytes) -> void* {
    void* p = ws + off;
    off += (bytes + 255) & ~(size_t)255;
    return p;
  };
  float*  deg  = (float*) alloc((size_t)N_NODES * 4);                 // [N]
  float*  agg  = (float*) alloc((size_t)N_NODES * HID_F * 4);         // reused: [N,128] then [N,256]
  __bf16* xbf  = (__bf16*)alloc((size_t)N_NODES * IN_F * 2);          // x in bf16
  __bf16* hn1  = (__bf16*)alloc((size_t)N_NODES * IN_F * 2);          // layer1 neighbor mean
  __bf16* hbf  = (__bf16*)alloc((size_t)N_NODES * HID_F * 2);         // relu(h1) in bf16
  __bf16* hn2  = (__bf16*)alloc((size_t)N_NODES * HID_F * 2);         // layer2 neighbor mean
  __bf16* wp1  = (__bf16*)alloc((size_t)(IN_F + IN_F) * HID_F * 2);   // packed [Wself1;Wneigh1]
  __bf16* wp2  = (__bf16*)alloc((size_t)(HID_F + HID_F) * OUT_F * 2); // packed [Wself2;Wneigh2]
  (void)ws_size; (void)in_sizes; (void)n_in; (void)out_size;

  const int T = 256;

  // ---- phase 0: zero deg + layer-1 agg, compute degrees, x->bf16, pack weights ----
  hipMemsetAsync(deg, 0, (size_t)N_NODES * 4, stream);
  hipMemsetAsync(agg, 0, (size_t)N_NODES * IN_F * 4, stream);
  k_degree<<<(N_EDGES + T - 1) / T, T, 0, stream>>>(dst, deg);
  k_f32_to_bf16<<<((size_t)N_NODES * IN_F + T - 1) / T, T, 0, stream>>>(x, xbf, N_NODES * IN_F);
  k_pack_w<<<(2 * IN_F * HID_F + T - 1) / T, T, 0, stream>>>(W_self1, W_neigh1,
                                                             IN_F, 2 * IN_F, HID_F, wp1);
  k_pack_w<<<(2 * HID_F * OUT_F + T - 1) / T, T, 0, stream>>>(W_self2, W_neigh2,
                                                              HID_F, 2 * HID_F, OUT_F, wp2);

  // ---- layer 1 aggregation: agg[dst,:] += x[src,:]  (E * 128/4 threads) ----
  {
    long long total = (long long)N_EDGES << 5;  // lgc=5 (128/4 chunks)
    k_scatter_f32<<<(unsigned)((total + T - 1) / T), T, 0, stream>>>(x, src, dst, agg, 5, IN_F);
  }
  k_neigh_mean<<<((size_t)N_NODES * IN_F + T - 1) / T, T, 0, stream>>>(
      agg, deg, hn1, 7, (long long)N_NODES * IN_F);

  // ---- layer 1 GEMM: h = relu([x|hn1] @ [Ws1;Wn1] + b1) -> hbf [N,256] ----
  {
    // 4 waves x 4 ntiles = all 16 N-tiles of HID_F per block
    dim3 grid(N_NODES / 16, 1);
    k_sage_gemm<4><<<grid, 128, 0, stream>>>(xbf, hn1, IN_F, IN_F, 2 * IN_F,
                                             wp1, b1, HID_F, nullptr, hbf, 1);
  }

  // ---- layer 2 aggregation: agg[dst,:] += h[src,:]  (E * 256/4 threads) ----
  hipMemsetAsync(agg, 0, (size_t)N_NODES * HID_F * 4, stream);
  {
    long long total = (long long)N_EDGES << 6;  // lgc=6 (256/4 chunks)
    k_scatter_bf16<<<(unsigned)((total + T - 1) / T), T, 0, stream>>>(hbf, src, dst, agg, 6, HID_F);
  }
  k_neigh_mean<<<((size_t)N_NODES * HID_F + T - 1) / T, T, 0, stream>>>(
      agg, deg, hn2, 8, (long long)N_NODES * HID_F);

  // ---- layer 2 GEMM: out = [h|hn2] @ [Ws2;Wn2] + b2 -> f32 d_out [N,128] ----
  {
    // 2 waves x 4 ntiles = all 8 N-tiles of OUT_F per block
    dim3 grid(N_NODES / 16, 1);
    k_sage_gemm<4><<<grid, 64, 0, stream>>>(hbf, hn2, HID_F, HID_F, 2 * HID_F,
                                            wp2, b2, OUT_F, out, nullptr, 0);
  }
}
